// MultiHeadAttention_11991548691070
// MI455X (gfx1250) — matrix operand order repaired and checked by reading, not compiled, not run on hardware
//
#include <hip/hip_runtime.h>
#include <hip/hip_bf16.h>

// ---------------------------------------------------------------------------
// MHA forward for B=2, T=2048, C=1024, H=16, D=64 on gfx1250 (MI455X).
// bf16 WMMA everywhere, double-buffered LDS tiles staged with
// GLOBAL_LOAD_ASYNC_TO_LDS_B128 when the toolchain exposes the builtins.
// ---------------------------------------------------------------------------

#define B_ 2
#define T_ 2048
#define C_ 1024
#define H_ 16
#define D_ 64
#define M_ (B_ * T_)

typedef __bf16 v16bf __attribute__((ext_vector_type(16)));
typedef float  v8f   __attribute__((ext_vector_type(8)));
typedef unsigned int u32x8 __attribute__((ext_vector_type(8)));

#if defined(__AMDGCN__) && __has_builtin(__builtin_amdgcn_global_load_async_to_lds_b128) && __has_builtin(__builtin_amdgcn_s_wait_asynccnt)
#define USE_ASYNC 1
#else
#define USE_ASYNC 0
#endif

__device__ __forceinline__ unsigned short to_bf16(float f) {
  unsigned int u = __float_as_uint(f);
  u += 0x7FFFu + ((u >> 16) & 1u);           // round-to-nearest-even
  return (unsigned short)(u >> 16);
}

__device__ __forceinline__ v8f wmma_bf16(const unsigned int a[8],
                                         const unsigned int b[8], v8f c) {
  u32x8 av, bv;
#pragma unroll
  for (int i = 0; i < 8; ++i) { av[i] = a[i]; bv[i] = b[i]; }
  return __builtin_amdgcn_wmma_f32_16x16x32_bf16(
      false, __builtin_bit_cast(v16bf, av),
      false, __builtin_bit_cast(v16bf, bv),
      (short)0, c, false, false);
}

// 16-byte global->LDS copy: async on CDNA5, sync fallback otherwise
#if USE_ASYNC
typedef int vsi4 __attribute__((vector_size(16)));
typedef __attribute__((address_space(1))) vsi4* gvec_t;
typedef __attribute__((address_space(3))) vsi4* lvec_t;
#endif

__device__ __forceinline__ void cp16(const unsigned short* g, unsigned short* l) {
#if USE_ASYNC
  __builtin_amdgcn_global_load_async_to_lds_b128(
      (gvec_t)(unsigned long long)(uintptr_t)g,
      (lvec_t)(unsigned int)(uintptr_t)l, 0, 0);
#else
  typedef unsigned int u32x4 __attribute__((ext_vector_type(4)));
  *(u32x4*)l = *(const u32x4*)g;
#endif
}

__device__ __forceinline__ void wait_async() {
#if USE_ASYNC
  __builtin_amdgcn_s_wait_asynccnt(0);
#endif
}

// cross-lane xor exchange (stays inside 16-lane half for mask < 16)
__device__ __forceinline__ float lane_xor(float v, int mask) {
  int lane = (int)(threadIdx.x & 31u);
  int idx = (lane ^ mask) << 2;
  return __int_as_float(__builtin_amdgcn_ds_bpermute(idx, __float_as_int(v)));
}

// ---------------------------------------------------------------------------
__global__ void k_f32_to_bf16(const float* __restrict__ in,
                              unsigned short* __restrict__ out, int n) {
  int i = blockIdx.x * blockDim.x + threadIdx.x;
  if (i < n) out[i] = to_bf16(in[i]);
}

// in f32 [K,N] -> out bf16 [N,K]  (one-time weight transpose for async staging)
__global__ void k_cvt_T(const float* __restrict__ in,
                        unsigned short* __restrict__ out, int K, int N) {
  int i = blockIdx.x * blockDim.x + threadIdx.x;
  if (i >= K * N) return;
  int k = i % K, n = i / K;
  out[(size_t)n * K + k] = to_bf16(in[(size_t)k * N + n]);
}

// qkv f32 [B,T,3C] -> Q,K bf16 [B,H,T,D]; V bf16 transposed [B,H,D,T].
// Q pre-scaled by 1/sqrt(D).
__global__ void k_qkv_split(const float* __restrict__ qkv,
                            unsigned short* __restrict__ Qb,
                            unsigned short* __restrict__ Kb,
                            unsigned short* __restrict__ Vt) {
  int i = blockIdx.x * blockDim.x + threadIdx.x;      // over B*T*C
  if (i >= B_ * T_ * C_) return;
  int d = i & (D_ - 1);
  int h = (i >> 6) & (H_ - 1);
  int t = (i >> 10) & (T_ - 1);
  int b = i >> 21;
  int bh = b * H_ + h;
  size_t src = (size_t)(b * T_ + t) * (3 * C_) + h * D_ + d;
  size_t dst = ((size_t)bh * T_ + t) * D_ + d;
  Qb[dst] = to_bf16(qkv[src] * 0.125f);
  Kb[dst] = to_bf16(qkv[src + C_]);
  Vt[((size_t)bh * D_ + d) * T_ + t] = to_bf16(qkv[src + 2 * C_]);
}

// ---------------------------------------------------------------------------
// C[M,N] f32 = A[M,K] bf16 @ Bt[N,K]^T bf16 + bias[N]
// block tile 128x128, 256 threads = 8 waves (4 m-slots x 2 n-slots),
// each wave computes 32x64; K staged in double-buffered 32-chunks.
__global__ __launch_bounds__(256) void k_gemm_bf16(
    const unsigned short* __restrict__ A, const unsigned short* __restrict__ Bt,
    const float* __restrict__ bias, float* __restrict__ Cout,
    int M, int N, int K) {
  __shared__ __align__(16) unsigned short As[2][128 * 32];   // [m][k]
  __shared__ __align__(16) unsigned short Bs[2][128 * 32];   // [n][k]
  const int tid = threadIdx.x;
  const int lane = tid & 31, wid = tid >> 5;
  const int hlf = lane >> 4, lrow = lane & 15;
  const int wm = wid & 3, wn = wid >> 2;
  const int mbase = blockIdx.y * 128;
  const int nbase = blockIdx.x * 128;

  // stage one 128x32 A tile + 128x32 B tile (1024 x 16B copies, 4/thread)
  auto stage = [&](int buf, int kc) {
#pragma unroll
    for (int it = 0; it < 2; ++it) {
      int i = tid + it * 256;
      int r = i >> 2, seg = (i & 3) * 8;
      cp16(&A[(size_t)(mbase + r) * K + kc + seg], &As[buf][r * 32 + seg]);
      cp16(&Bt[(size_t)(nbase + r) * K + kc + seg], &Bs[buf][r * 32 + seg]);
    }
  };

  v8f zero8 = {0.f, 0.f, 0.f, 0.f, 0.f, 0.f, 0.f, 0.f};
  v8f acc[2][4] = {{zero8, zero8, zero8, zero8}, {zero8, zero8, zero8, zero8}};

  stage(0, 0);
  int cur = 0;
  for (int kc = 0; kc < K; kc += 32, cur ^= 1) {
    wait_async();
    __syncthreads();                 // tile ready; all waves done with buffers
    if (kc + 32 < K) stage(cur ^ 1, kc + 32);

    unsigned int af[2][8];
#pragma unroll
    for (int ms = 0; ms < 2; ++ms) {
      const unsigned short* arow = &As[cur][(wm * 32 + ms * 16 + lrow) * 32];
#pragma unroll
      for (int p = 0; p < 8; ++p) {
        int pair = p + hlf * 8;
        af[ms][p] = *(const unsigned int*)&arow[2 * pair];
      }
    }
#pragma unroll
    for (int ns = 0; ns < 4; ++ns) {
      unsigned int bfr[8];
      const unsigned short* bcol = &Bs[cur][(wn * 64 + ns * 16 + lrow) * 32];
#pragma unroll
      for (int p = 0; p < 8; ++p)
        bfr[p] = *(const unsigned int*)&bcol[2 * (p + hlf * 8)];
#pragma unroll
      for (int ms = 0; ms < 2; ++ms)
        acc[ms][ns] = wmma_bf16(af[ms], bfr, acc[ms][ns]);
    }
  }

#pragma unroll
  for (int ns = 0; ns < 4; ++ns) {
    int col = nbase + wn * 64 + ns * 16 + lrow;
    float bv = bias[col];
#pragma unroll
    for (int ms = 0; ms < 2; ++ms)
#pragma unroll
      for (int j = 0; j < 8; ++j) {
        int row = mbase + wm * 32 + ms * 16 + hlf * 8 + j;
        Cout[(size_t)row * N + col] = acc[ms][ns][j] + bv;
      }
  }
}

// ---------------------------------------------------------------------------
// Flash attention: grid (T/64, B*H), 128 threads = 4 waves; each wave owns a
// 16-row Q tile, streams causal 64-key K/V tiles (double-buffered) thru LDS.
__global__ __launch_bounds__(128) void k_flash(
    const unsigned short* __restrict__ Qb, const unsigned short* __restrict__ Kb,
    const unsigned short* __restrict__ Vt, const int* __restrict__ tmask,
    unsigned short* __restrict__ Oc /* bf16 [B,T,C] */) {
  __shared__ __align__(16) unsigned short Ks[2][64 * 64];    // [key][d]
  __shared__ __align__(16) unsigned short Vts[2][64 * 64];   // [d][key]
  __shared__ __align__(16) unsigned short Ps[4][16 * 64];    // per-wave P
  __shared__ float mb[2][64];

  const int tid = threadIdx.x;
  const int lane = tid & 31, wid = tid >> 5;
  const int hlf = lane >> 4, lrow = lane & 15;
  const int bh = blockIdx.y;
  const int b = bh / H_, h = bh % H_;
  const int qb = blockIdx.x;
  const int qr0 = qb * 64 + wid * 16;
  const size_t baseK = (size_t)bh * T_ * D_;
  const size_t baseV = (size_t)bh * D_ * T_;

  // stage 64x64 K tile ([key][d]) + 64x64 V tile ([d][key]), 8 x 16B / thread
  auto stage_kv = [&](int buf, int j0) {
#pragma unroll
    for (int it = 0; it < 4; ++it) {
      int i = tid + it * 128;
      int r = i >> 3, seg = (i & 7) * 8;
      cp16(&Kb[baseK + (size_t)(j0 + r) * D_ + seg], &Ks[buf][r * 64 + seg]);
      cp16(&Vt[baseV + (size_t)r * T_ + j0 + seg], &Vts[buf][r * 64 + seg]);
    }
    if (tid < 64) mb[buf][tid] = tmask[b * T_ + j0 + tid] ? 0.f : -1e30f;
  };

  // Q fragments (2 d-chunks of 32), straight from global
  unsigned int qf[2][8];
  {
    const unsigned short* qrow = &Qb[baseK + (size_t)(qr0 + lrow) * D_];
#pragma unroll
    for (int c = 0; c < 2; ++c)
#pragma unroll
      for (int p = 0; p < 8; ++p) {
        int pair = p + hlf * 8;
        qf[c][p] = *(const unsigned int*)&qrow[c * 32 + 2 * pair];
      }
  }

  v8f zero8 = {0.f, 0.f, 0.f, 0.f, 0.f, 0.f, 0.f, 0.f};
  v8f oacc[4] = {zero8, zero8, zero8, zero8};
  float mrun[8], lrun[8];
#pragma unroll
  for (int j = 0; j < 8; ++j) { mrun[j] = -1e30f; lrun[j] = 0.f; }

  const int kend = qb * 64 + 64;     // causal: keys < (q-tile end)
  stage_kv(0, 0);
  int cur = 0;
  for (int j0 = 0; j0 < kend; j0 += 64, cur ^= 1) {
    wait_async();
    __syncthreads();
    if (j0 + 64 < kend) stage_kv(cur ^ 1, j0 + 64);

    // S = Q @ K^T  (contraction over d)
    v8f sacc[4] = {zero8, zero8, zero8, zero8};
#pragma unroll
    for (int ns = 0; ns < 4; ++ns) {
      const unsigned short* bcol = &Ks[cur][(ns * 16 + lrow) * 64];
#pragma unroll
      for (int c = 0; c < 2; ++c) {
        unsigned int bfr[8];
#pragma unroll
        for (int p = 0; p < 8; ++p)
          bfr[p] = *(const unsigned int*)&bcol[c * 32 + 2 * (p + hlf * 8)];
        sacc[ns] = wmma_bf16(qf[c], bfr, sacc[ns]);
      }
    }

    // online softmax (rows live across 16-lane halves in C layout)
    float pbuf[4][8];
#pragma unroll
    for (int j = 0; j < 8; ++j) {
      int row = qr0 + hlf * 8 + j;
      float mx = -1e30f;
#pragma unroll
      for (int ns = 0; ns < 4; ++ns) {
        int col = j0 + ns * 16 + lrow;
        float s = sacc[ns][j] + mb[cur][ns * 16 + lrow];
        if (col > row) s = -1e30f;             // causal
        pbuf[ns][j] = s;
        mx = fmaxf(mx, s);
      }
      float v = mx;
#pragma unroll
      for (int m = 1; m < 16; m <<= 1) v = fmaxf(v, lane_xor(v, m));
      float mnew = fmaxf(mrun[j], v);
      float alpha = __expf(mrun[j] - mnew);
      mrun[j] = mnew;
      float rs = 0.f;
#pragma unroll
      for (int ns = 0; ns < 4; ++ns) {
        float p = __expf(pbuf[ns][j] - mnew);
        pbuf[ns][j] = p;
        rs += p;
      }
#pragma unroll
      for (int m = 1; m < 16; m <<= 1) rs += lane_xor(rs, m);
      lrun[j] = lrun[j] * alpha + rs;
#pragma unroll
      for (int ns = 0; ns < 4; ++ns) oacc[ns][j] *= alpha;
    }

    // re-layout P (C-format) -> LDS -> A-format fragments
    unsigned short* ps = Ps[wid];
#pragma unroll
    for (int ns = 0; ns < 4; ++ns)
#pragma unroll
      for (int j = 0; j < 8; ++j)
        ps[(hlf * 8 + j) * 64 + ns * 16 + lrow] = to_bf16(pbuf[ns][j]);

    // O += P @ V  (contraction over key)
#pragma unroll
    for (int c = 0; c < 2; ++c) {
      unsigned int af[8];
      const unsigned short* prow = &ps[lrow * 64];
#pragma unroll
      for (int p = 0; p < 8; ++p) {
        int pair = p + hlf * 8;
        af[p] = *(const unsigned int*)&prow[c * 32 + 2 * pair];
      }
#pragma unroll
      for (int ns = 0; ns < 4; ++ns) {
        unsigned int bfr[8];
        const unsigned short* vcol = &Vts[cur][(ns * 16 + lrow) * 64];
#pragma unroll
        for (int p = 0; p < 8; ++p)
          bfr[p] = *(const unsigned int*)&vcol[c * 32 + 2 * (p + hlf * 8)];
        oacc[ns] = wmma_bf16(af, bfr, oacc[ns]);
      }
    }
  }

  // normalize, write bf16 into [B,T,H*D] for the projection GEMM
#pragma unroll
  for (int ns = 0; ns < 4; ++ns)
#pragma unroll
    for (int j = 0; j < 8; ++j) {
      int row = qr0 + hlf * 8 + j;
      int dcol = ns * 16 + lrow;
      float o = oacc[ns][j] / lrun[j];
      Oc[(size_t)(b * T_ + row) * C_ + h * D_ + dcol] = to_bf16(o);
    }
}

// ---------------------------------------------------------------------------
extern "C" void kernel_launch(void* const* d_in, const int* in_sizes, int n_in,
                              void* d_out, int out_size, void* d_ws, size_t ws_size,
                              hipStream_t stream) {
  const float* x      = (const float*)d_in[0];
  const int*   tmask  = (const int*)d_in[1];
  const float* W_attn = (const float*)d_in[2];
  const float* b_attn = (const float*)d_in[3];
  const float* W_proj = (const float*)d_in[4];
  const float* b_proj = (const float*)d_in[5];
  float* out = (float*)d_out;

  // workspace layout (~99 MB total)
  char* ws = (char*)d_ws;
  unsigned short* x_bf  = (unsigned short*)ws;  ws += (size_t)M_ * C_ * 2;
  unsigned short* Wa_t  = (unsigned short*)ws;  ws += (size_t)3 * C_ * C_ * 2;
  unsigned short* Wp_t  = (unsigned short*)ws;  ws += (size_t)C_ * C_ * 2;
  float*          qkv   = (float*)ws;           ws += (size_t)M_ * 3 * C_ * 4;
  unsigned short* Qb    = (unsigned short*)ws;  ws += (size_t)M_ * C_ * 2;
  unsigned short* Kb    = (unsigned short*)ws;  ws += (size_t)M_ * C_ * 2;
  unsigned short* Vt    = (unsigned short*)ws;  ws += (size_t)M_ * C_ * 2;
  unsigned short* attn  = (unsigned short*)ws;  ws += (size_t)M_ * C_ * 2;

  int n;
  n = M_ * C_;
  k_f32_to_bf16<<<(n + 255) / 256, 256, 0, stream>>>(x, x_bf, n);
  n = C_ * 3 * C_;
  k_cvt_T<<<(n + 255) / 256, 256, 0, stream>>>(W_attn, Wa_t, C_, 3 * C_);
  n = C_ * C_;
  k_cvt_T<<<(n + 255) / 256, 256, 0, stream>>>(W_proj, Wp_t, C_, C_);

  // qkv = x @ W_attn + b_attn
  k_gemm_bf16<<<dim3((3 * C_) / 128, M_ / 128), 256, 0, stream>>>(
      x_bf, Wa_t, b_attn, qkv, M_, 3 * C_, C_);

  n = B_ * T_ * C_;
  k_qkv_split<<<(n + 255) / 256, 256, 0, stream>>>(qkv, Qb, Kb, Vt);

  k_flash<<<dim3(T_ / 64, B_ * H_), 128, 0, stream>>>(Qb, Kb, Vt, tmask, attn);

  // out = attn @ W_proj + b_proj
  k_gemm_bf16<<<dim3(C_ / 128, M_ / 128), 256, 0, stream>>>(
      attn, Wp_t, b_proj, out, M_, C_, C_);
}
